// Predictor_62697932587670
// MI455X (gfx1250) — matrix operand order repaired and checked by reading, not compiled, not run on hardware
//
#include <hip/hip_runtime.h>
#include <hip/hip_bf16.h>
#include <stdint.h>

#define BATCH    16384
#define DIM      512
#define NCLS     3992
#define NCPAD    4096
#define KFLAT    3072   // 6*512
#define KHALF    1536   // 3*512

typedef __attribute__((ext_vector_type(16))) __bf16 v16bf;
typedef __attribute__((ext_vector_type(8)))  float  v8f;

// ---------------------------------------------------------------- CDNA5 async global->LDS
// LDS byte offset = low 32 bits of flat pointer to __shared__ (ISA 10.2).
__device__ __forceinline__ uint32_t lds_off(const void* p) {
    return (uint32_t)(uintptr_t)p;
}
__device__ __forceinline__ void async_ld_b128(uint32_t ldsaddr, const void* gaddr) {
    asm volatile("global_load_async_to_lds_b128 %0, %1, off"
                 :: "v"(ldsaddr), "v"((uint64_t)(uintptr_t)gaddr)
                 : "memory");
}
__device__ __forceinline__ void wait_async0() {
    asm volatile("s_wait_asynccnt 0" ::: "memory");
}

// ---------------------------------------------------------------- weight converts (f32 [K,N] -> bf16 [N,K] transposed)
__global__ void k_cvt_t(const float* __restrict__ in, __bf16* __restrict__ out, int K, int N) {
    int i = blockIdx.x * blockDim.x + threadIdx.x;
    if (i < K * N) {
        int n = i / K, k = i - n * K;
        out[i] = (__bf16)in[k * N + n];
    }
}
// e3_W1 with x3 scaling folded in: input feature k scaled by (k<split ? a : b)
__global__ void k_cvt_t_scale(const float* __restrict__ in, __bf16* __restrict__ out,
                              const float* __restrict__ sa, const float* __restrict__ sb,
                              int K, int N, int split) {
    int i = blockIdx.x * blockDim.x + threadIdx.x;
    if (i < K * N) {
        int n = i / K, k = i - n * K;
        float s = (k < split) ? sa[0] : sb[0];
        out[i] = (__bf16)(in[k * N + n] * s);
    }
}
// dec_W2 [512,3992] -> transposed + zero padded [4096,512]
__global__ void k_cvt_t_pad(const float* __restrict__ in, __bf16* __restrict__ out,
                            int K, int N, int Npad) {
    int i = blockIdx.x * blockDim.x + threadIdx.x;
    if (i < Npad * K) {
        int n = i / K, k = i - n * K;
        out[i] = (n < N) ? (__bf16)in[k * N + n] : (__bf16)0.0f;
    }
}

// fusion_hs [L,B,D] f32 -> flat [B, L*D] bf16, 8 elems/thread
__global__ void k_pack_flat(const float* __restrict__ fusion, __bf16* __restrict__ flat) {
    int t = blockIdx.x * blockDim.x + threadIdx.x;   // < BATCH*KFLAT/8
    int i = t << 3;
    int b = i / KFLAT;
    int c = i - b * KFLAT;
    int l = c >> 9, d = c & 511;
    const float* src = fusion + (l * BATCH + b) * DIM + d;   // fits in int (<51M)
    __bf16 v[8];
#pragma unroll
    for (int j = 0; j < 8; ++j) v[j] = (__bf16)src[j];
    *(uint4*)(flat + i) = *(const uint4*)v;
}

// ---------------------------------------------------------------- gate (hard straight-through argmax)
__global__ void k_gate(const float* __restrict__ fusion,
                       const float* __restrict__ gW, const float* __restrict__ gb,
                       int* __restrict__ gidx) {
    int gt   = blockIdx.x * blockDim.x + threadIdx.x;
    int row  = gt >> 5;
    int lane = gt & 31;
    if (row >= BATCH) return;
    float a0 = 0.f, a1 = 0.f, a2 = 0.f, a3 = 0.f;
    for (int k = lane; k < KFLAT; k += 32) {
        int l = k >> 9, d = k & 511;
        float x = fusion[(l * BATCH + row) * DIM + d];
        const float* w = gW + k * 4;
        a0 += x * w[0]; a1 += x * w[1]; a2 += x * w[2]; a3 += x * w[3];
    }
    for (int off = 16; off > 0; off >>= 1) {
        a0 += __shfl_xor(a0, off, 32);
        a1 += __shfl_xor(a1, off, 32);
        a2 += __shfl_xor(a2, off, 32);
        a3 += __shfl_xor(a3, off, 32);
    }
    if (lane == 0) {
        float v0 = a0 + gb[0], v1 = a1 + gb[1], v2 = a2 + gb[2], v3 = a3 + gb[3];
        int best = 0; float bv = v0;
        if (v1 > bv) { bv = v1; best = 1; }
        if (v2 > bv) { bv = v2; best = 2; }
        if (v3 > bv) { bv = v3; best = 3; }
        gidx[row] = best;
    }
}

// ---------------------------------------------------------------- WMMA GEMM
// C[M,N] = act(A[M,K] @ Wt[N,K]^T + bias).  A bf16 [M,K] row-major, Wt bf16 [N,K] row-major.
// Block tile 128x128, 8 waves (4m x 2n), wave tile 32x64 = 2x4 wmma 16x16x32.
// Double-buffered LDS filled with global_load_async_to_lds_b128 (ASYNCcnt pipeline),
// last iteration peeled so the steady-state loop is branch-free.
// ACT: 0=none 1=relu 2=sigmoid.  OUTBF: 1 -> bf16, 0 -> f32.
#define AK 40                       // padded K-stride in elements (80B, multiple of 16B)
#define TILE_BYTES (128 * AK * 2)   // one LDS buffer: 10240 B

template<int ACT, int OUTBF>
__global__ __launch_bounds__(256)
void k_gemm(const __bf16* __restrict__ A, int lda,
            const __bf16* __restrict__ Wt, int ldw,
            const float* __restrict__ bias,
            void* __restrict__ Cout, int ldc, int Nreal, int K) {
    __shared__ __bf16 As[2][128 * AK];
    __shared__ __bf16 Bs[2][128 * AK];

    const int tid    = threadIdx.x;
    const int lane   = tid & 31;
    const int wid    = tid >> 5;
    const int wm     = wid >> 1;            // 0..3 -> 32-row slice
    const int wn     = wid & 1;             // 0..1 -> 64-col slice
    const int blockM = blockIdx.y * 128;
    const int blockN = blockIdx.x * 128;

    // per-thread async-copy slots: chunk rows (tid>>2, tid>>2 + 64), 8 bf16 at col (tid&3)*8
    const int crow = tid >> 2;
    const int ckk  = (tid & 3) << 3;
    const __bf16* pA0 = A  + (blockM + crow) * lda + ckk;
    const __bf16* pA1 = pA0 + 64 * lda;
    const __bf16* pB0 = Wt + (blockN + crow) * ldw + ckk;
    const __bf16* pB1 = pB0 + 64 * ldw;
    const uint32_t aL = lds_off(As) + (uint32_t)(crow * AK + ckk) * 2;
    const uint32_t bL = lds_off(Bs) + (uint32_t)(crow * AK + ckk) * 2;
    const uint32_t rowHop = (uint32_t)(64 * AK * 2);

    v8f zero = {0.f, 0.f, 0.f, 0.f, 0.f, 0.f, 0.f, 0.f};
    v8f acc[2][4];
#pragma unroll
    for (int i = 0; i < 2; ++i)
#pragma unroll
        for (int j = 0; j < 4; ++j) acc[i][j] = zero;

    const int hl  = lane >> 4;
    const int l15 = lane & 15;
    const int nK  = K >> 5;

    auto compute = [&](int cur) {
        v16bf afrag[2], bfrag[4];
#pragma unroll
        for (int i = 0; i < 2; ++i) {
            const __bf16* p = &As[cur][(wm * 32 + i * 16 + l15) * AK];
            uint4* f = (uint4*)&afrag[i];
            f[0] = *(const uint4*)(p + hl * 8);        // K 0..7  / 8..15
            f[1] = *(const uint4*)(p + 16 + hl * 8);   // K 16..23 / 24..31
        }
#pragma unroll
        for (int j = 0; j < 4; ++j) {
            const __bf16* p = &Bs[cur][(wn * 64 + j * 16 + l15) * AK];
            uint4* f = (uint4*)&bfrag[j];
            f[0] = *(const uint4*)(p + hl * 8);
            f[1] = *(const uint4*)(p + 16 + hl * 8);
        }
#pragma unroll
        for (int i = 0; i < 2; ++i)
#pragma unroll
            for (int j = 0; j < 4; ++j)
                acc[i][j] = __builtin_amdgcn_wmma_f32_16x16x32_bf16(
                    false, afrag[i], false, bfrag[j], (short)0, acc[i][j], false, false);
    };

    // prime buffer 0 with K-tile 0
    async_ld_b128(aL,          pA0);
    async_ld_b128(aL + rowHop, pA1);
    async_ld_b128(bL,          pB0);
    async_ld_b128(bL + rowHop, pB1);
    wait_async0();
    __syncthreads();

    // steady state: branch-free; fill buf[kt+1] while computing buf[kt]
    for (int kt = 0; kt < nK - 1; ++kt) {
        pA0 += 32; pA1 += 32; pB0 += 32; pB1 += 32;
        const uint32_t bofs = (uint32_t)((kt + 1) & 1) * TILE_BYTES;
        async_ld_b128(aL + bofs,          pA0);
        async_ld_b128(aL + bofs + rowHop, pA1);
        async_ld_b128(bL + bofs,          pB0);
        async_ld_b128(bL + bofs + rowHop, pB1);

        compute(kt & 1);

        wait_async0();      // this iteration's fills landed
        __syncthreads();    // everyone done reading buf[kt&1]
    }
    compute((nK - 1) & 1);  // peeled last tile, no fill / wait / barrier

    // epilogue: C/D layout -> col = l15, VGPR r -> row r + hl*8
#pragma unroll
    for (int i = 0; i < 2; ++i) {
#pragma unroll
        for (int j = 0; j < 4; ++j) {
            int n = blockN + wn * 64 + j * 16 + l15;
            if (n >= Nreal) continue;
            float bv = bias[n];
#pragma unroll
            for (int r = 0; r < 8; ++r) {
                int m = blockM + wm * 32 + i * 16 + hl * 8 + r;
                float v = acc[i][j][r] + bv;
                if (ACT == 1) v = v > 0.f ? v : 0.f;
                if (ACT == 2) v = 1.f / (1.f + __expf(-v));
                if (OUTBF) ((__bf16*)Cout)[m * ldc + n] = (__bf16)v;
                else       ((float*)Cout)[m * ldc + n]  = v;
            }
        }
    }
}

// ---------------------------------------------------------------- hard select (8 bf16 / thread)
__global__ void k_select(const __bf16* __restrict__ eout, const int* __restrict__ gidx,
                         __bf16* __restrict__ sel) {
    int t = blockIdx.x * blockDim.x + threadIdx.x;   // < BATCH*DIM/8
    int i = t << 3;
    int b = i >> 9;
    int g = gidx[b];
    *(uint4*)(sel + i) = *(const uint4*)(eout + g * (BATCH * DIM) + i);
}

// ---------------------------------------------------------------- host
static void launch_gemm(int act, int outbf, hipStream_t s,
                        const __bf16* A, int lda, const __bf16* Wt, int ldw,
                        const float* bias, void* C, int ldc, int Nreal, int Npad, int K) {
    dim3 grid(Npad / 128, BATCH / 128);
    if (outbf) {
        if (act == 1)      k_gemm<1, 1><<<grid, 256, 0, s>>>(A, lda, Wt, ldw, bias, C, ldc, Nreal, K);
        else if (act == 2) k_gemm<2, 1><<<grid, 256, 0, s>>>(A, lda, Wt, ldw, bias, C, ldc, Nreal, K);
        else               k_gemm<0, 1><<<grid, 256, 0, s>>>(A, lda, Wt, ldw, bias, C, ldc, Nreal, K);
    } else {
        k_gemm<0, 0><<<grid, 256, 0, s>>>(A, lda, Wt, ldw, bias, C, ldc, Nreal, K);
    }
}

extern "C" void kernel_launch(void* const* d_in, const int* in_sizes, int n_in,
                              void* d_out, int out_size, void* d_ws, size_t ws_size,
                              hipStream_t stream) {
    (void)in_sizes; (void)n_in; (void)out_size; (void)ws_size;
    const float* fusion = (const float*)d_in[0];
    const float* gate_W = (const float*)d_in[1];
    const float* gate_b = (const float*)d_in[2];
    const float* eW1[4] = {(const float*)d_in[3],  (const float*)d_in[7],
                           (const float*)d_in[11], (const float*)d_in[15]};
    const float* eB1[4] = {(const float*)d_in[4],  (const float*)d_in[8],
                           (const float*)d_in[12], (const float*)d_in[16]};
    const float* eW2[4] = {(const float*)d_in[5],  (const float*)d_in[9],
                           (const float*)d_in[13], (const float*)d_in[17]};
    const float* eB2[4] = {(const float*)d_in[6],  (const float*)d_in[10],
                           (const float*)d_in[14], (const float*)d_in[18]};
    const float* e3_a  = (const float*)d_in[19];
    const float* e3_b  = (const float*)d_in[20];
    const float* decW1 = (const float*)d_in[21];
    const float* decB1 = (const float*)d_in[22];
    const float* decW2 = (const float*)d_in[23];
    const float* decB2 = (const float*)d_in[24];

    // workspace carve-up
    char* w = (char*)d_ws;
    auto take = [&](size_t bytes) { void* p = (void*)w; w += (bytes + 255) & ~(size_t)255; return p; };
    __bf16* flat = (__bf16*)take((size_t)BATCH * KFLAT * 2);        // [B,3072]
    __bf16* w1t[4];                                                  // transposed [N=512, K]
    w1t[0] = (__bf16*)take((size_t)DIM * KHALF * 2);
    w1t[1] = (__bf16*)take((size_t)DIM * KHALF * 2);
    w1t[2] = (__bf16*)take((size_t)DIM * KFLAT * 2);
    w1t[3] = (__bf16*)take((size_t)DIM * KFLAT * 2);
    __bf16* w2t[4];                                                  // [512,512]
    for (int e = 0; e < 4; ++e) w2t[e] = (__bf16*)take((size_t)DIM * DIM * 2);
    __bf16* dW1t = (__bf16*)take((size_t)DIM * DIM * 2);             // [512,512]
    __bf16* dW2t = (__bf16*)take((size_t)NCPAD * DIM * 2);           // [4096,512] padded
    int*    gidx = (int*)take((size_t)BATCH * 4);
    __bf16* htmp = (__bf16*)take((size_t)BATCH * DIM * 2);           // reused hidden
    __bf16* eout = (__bf16*)take((size_t)4 * BATCH * DIM * 2);       // [4,B,512]
    __bf16* sel  = (__bf16*)take((size_t)BATCH * DIM * 2);

    auto cvt_t = [&](const float* in, __bf16* out, int K, int N) {
        k_cvt_t<<<(K * N + 255) / 256, 256, 0, stream>>>(in, out, K, N);
    };
    // weight conversion + transpose (e3_W1 absorbs x3 scaling)
    cvt_t(eW1[0], w1t[0], KHALF, DIM);
    cvt_t(eW1[1], w1t[1], KHALF, DIM);
    cvt_t(eW1[2], w1t[2], KFLAT, DIM);
    k_cvt_t_scale<<<(KFLAT * DIM + 255) / 256, 256, 0, stream>>>(
        eW1[3], w1t[3], e3_a, e3_b, KFLAT, DIM, KHALF);
    for (int e = 0; e < 4; ++e) cvt_t(eW2[e], w2t[e], DIM, DIM);
    cvt_t(decW1, dW1t, DIM, DIM);
    k_cvt_t_pad<<<(NCPAD * DIM + 255) / 256, 256, 0, stream>>>(decW2, dW2t, DIM, NCLS, NCPAD);

    // activations + gate
    k_pack_flat<<<(BATCH * KFLAT / 8) / 256, 256, 0, stream>>>(fusion, flat);
    k_gate<<<(BATCH * 32) / 256, 256, 0, stream>>>(fusion, gate_W, gate_b, gidx);

    // experts: h = relu(X @ W1 + b1) ; out_e = h @ W2 + b2
    const __bf16* Ae[4] = {flat, flat + KHALF, flat, flat};
    const int     Ke[4] = {KHALF, KHALF, KFLAT, KFLAT};
    for (int e = 0; e < 4; ++e) {
        launch_gemm(1, 1, stream, Ae[e], KFLAT, w1t[e], Ke[e], eB1[e],
                    htmp, DIM, DIM, DIM, Ke[e]);
        launch_gemm(0, 1, stream, htmp, DIM, w2t[e], DIM, eB2[e],
                    eout + (size_t)e * BATCH * DIM, DIM, DIM, DIM, DIM);
    }

    // hard straight-through gate == select argmax expert
    k_select<<<(BATCH * DIM / 8) / 256, 256, 0, stream>>>(eout, gidx, sel);

    // decoder: sigmoid(sel @ dW1 + b1) @ dW2 + b2 -> f32 [B, 3992]
    launch_gemm(2, 1, stream, sel, DIM, dW1t, DIM, decB1, htmp, DIM, DIM, DIM, DIM);
    launch_gemm(0, 0, stream, htmp, DIM, dW2t, DIM, decB2, d_out, NCLS, NCLS, NCPAD, DIM);
}
